// CliffordPC_84164179132821
// MI455X (gfx1250) — matrix operand order repaired and checked by reading, not compiled, not run on hardware
//
#include <hip/hip_runtime.h>
#include <stdint.h>

// ---------------------------------------------------------------------------
// Clifford predictive coding as 80 sequential bf16-WMMA GEMMs (f32 accumulate)
// Cl(3,0): K = 8 blades. The sparse Cayley contraction is folded into
// pre-expanded signed weight matrices (stored column-major, bf16) so each
// geometric product is a plain GEMM on the v_wmma_f32_16x16x32_bf16 pipe.
// B tiles are staged into LDS with CDNA5 async copies (ASYNCcnt), double
// buffered. Padded-row LDS layout (stride % 256 == 16) gives bank-conflict-
// free fragment reads with purely affine addresses (immediate ds offsets),
// so the scheduler can clause loads and pipeline them across WMMAs.
// ---------------------------------------------------------------------------

#define KB     8
#define BATCH  64
#define D0     256
#define D1     512
#define F0     (D0 * KB)        // 2048 flat features (layers 0 and 2)
#define F1     (D1 * KB)        // 4096 flat features (layer 1)
#define S0     (BATCH * F0)     // 131072
#define S1     (BATCH * F1)     // 262144
#define N_ITER 20
#define ALPHA  0.1f

#define KC       512                    // K-chunk staged per LDS buffer
#define CPR      (KC / 8)               // 16-byte chunks per row = 64
#define ROWB     (KC * 2 + 16)          // padded row stride = 1040 B
#define BUFBYTES (16 * ROWB)            // 16.25 KB per buffer
#define LDSBYTES (2 * BUFBYTES)         // 32.5 KB double buffer

typedef __attribute__((ext_vector_type(16))) __bf16 v16bf;
typedef __attribute__((ext_vector_type(8)))  __bf16 v8bf;
typedef __attribute__((ext_vector_type(8)))  float  v8f;

// ---------------------------------------------------------------------------
// CDNA5 async copy global->LDS (tracked by ASYNCcnt), per-lane LDS address.
// ---------------------------------------------------------------------------
__device__ __forceinline__ void async_load_b128(unsigned lds_addr, const void* gaddr) {
    asm volatile("global_load_async_to_lds_b128 %0, %1, off"
                 :: "v"(lds_addr), "v"(gaddr) : "memory");
}
__device__ __forceinline__ void wait_async0() {
    asm volatile("s_wait_asynccnt 0" ::: "memory");
}

// ---------------------------------------------------------------------------
// Weight expansion:  out[(m*8+k) * (Ndim*8) + (n*8+a)] =
//     sign(a, a^k) * (rev ? REV[a^k] : 1) * w[m*sm + n*sn + (a^k)]
// Stored "column-major" w.r.t. the GEMM (row = output feature q, contiguous =
// reduction feature p) so a WMMA B-fragment is two contiguous 16-byte runs.
// ---------------------------------------------------------------------------
__global__ void expand_weights(const float* __restrict__ w,
                               __bf16* __restrict__ out,
                               int Ndim, int sm, int sn, int do_rev) {
    int i = blockIdx.x * blockDim.x + threadIdx.x;   // over Mdim*8 * Ndim*8
    int P = Ndim * KB;
    int q = i / P;
    int p = i - q * P;
    int m = q >> 3, k = q & 7;
    int n = p >> 3, a = p & 7;
    int c = a ^ k;
    // canonical reordering sign for e_a * e_c (metric is all +1)
    int s = 0, aa = a >> 1;
    while (aa) { s += __popc(aa & c); aa >>= 1; }
    float sgn = (s & 1) ? -1.0f : 1.0f;
    if (do_rev) {                       // reversion sign (-1)^{r(r-1)/2}
        int r = __popc(c);
        if (((r * (r - 1)) / 2) & 1) sgn = -sgn;
    }
    out[i] = (__bf16)(sgn * w[m * sm + n * sn + c]);
}

// states0 = x ; states1 = states2 = 0 ; bf16 shadows = 0
__global__ void init_states(const float* __restrict__ x,
                            float* __restrict__ o0, float* __restrict__ o1,
                            float* __restrict__ o2,
                            __bf16* __restrict__ sb1, __bf16* __restrict__ sb2) {
    int i = blockIdx.x * blockDim.x + threadIdx.x;   // grid covers S1
    if (i < S0) { o0[i] = x[i]; o2[i] = 0.0f; sb2[i] = (__bf16)0.0f; }
    o1[i] = 0.0f;
    sb1[i] = (__bf16)0.0f;
}

// ---------------------------------------------------------------------------
// Stage one 16 x KC B-subtile (rows n0..n0+15 of column-major E) into the LDS
// buffer at byte offset bufbase, padded-row layout. 128 threads x 8 async
// b128 ops; consecutive tids take consecutive 16B chunks (coalesced runs).
// ---------------------------------------------------------------------------
__device__ __forceinline__ void stage_chunk(const __bf16* __restrict__ Ecm,
                                            int Kdim, int n0, int kbase,
                                            char* smem, unsigned bufbase) {
    unsigned sbase = (unsigned)(uintptr_t)smem + bufbase;
    for (int j = threadIdx.x; j < 16 * CPR; j += 128) {
        int rr = j >> 6;                 // j / CPR   (CPR == 64)
        int ck = j & (CPR - 1);
        const __bf16* g = Ecm + (size_t)(n0 + rr) * Kdim + kbase + (ck << 3);
        unsigned lds = sbase + (unsigned)(rr * ROWB) + (unsigned)(ck << 4);
        async_load_b128(lds, g);
    }
}

// ---------------------------------------------------------------------------
// One 16x16 f32 tile per wave:  C(16x16) += A(16x32 bf16) * B(32x16 bf16)
// A row-major (BATCH x Kdim) from global; B from padded LDS double buffer.
// Fragment layouts follow CDNA5 ISA 7.12.2 (wave32):
//   A: elems 0-7 -> K = k0 + half*8 .. ; elems 8-15 -> K = k0+16+half*8 ..
//   B: lane (half,r): elems 0-7 -> K = k0 + half*8 .. ; elems 8-15 -> K = k0+16+half*8 .. , N = r
//      -> LDS bytes [r*ROWB + half*16 + 2*k0, +16) and [+32, +48)   (affine, 16B aligned)
//   C/D: VGPR v, lane (half,r): M = half*8 + v, N = r
// ---------------------------------------------------------------------------
__device__ __forceinline__ v8f gemm_tile(const __bf16* __restrict__ Abf,
                                         const __bf16* __restrict__ Ecm,
                                         char* smem, int Kdim,
                                         int mb, int n0, int half, int r) {
    v8f acc = {};
    const __bf16* arow = Abf + (size_t)(mb + r) * Kdim + half * 8;
    const int nChunks = Kdim / KC;

    stage_chunk(Ecm, Kdim, n0, 0, smem, 0);
    wait_async0();
    __syncthreads();

    for (int c = 0; c < nChunks; ++c) {
        if (c + 1 < nChunks)   // prefetch next chunk while computing this one
            stage_chunk(Ecm, Kdim, n0, (c + 1) * KC, smem,
                        ((c + 1) & 1) ? BUFBYTES : 0u);
        const char* brow = smem + ((c & 1) ? BUFBYTES : 0) + r * ROWB + half * 16;
        const __bf16* arowc = arow + c * KC;
#pragma unroll
        for (int k0 = 0; k0 < KC; k0 += 32) {
            union { v16bf v; v8bf h[2]; } af, bfr;
            af.h[0] = *(const v8bf*)(arowc + k0);        // K = k0+half*8 ..
            af.h[1] = *(const v8bf*)(arowc + k0 + 16);   // K = k0+16+half*8 ..
            bfr.h[0] = *(const v8bf*)(brow + 2 * k0);        // K = k0+half*8 ..
            bfr.h[1] = *(const v8bf*)(brow + 2 * k0 + 32);
            acc = __builtin_amdgcn_wmma_f32_16x16x32_bf16(
                      false, af.v, false, bfr.v, (short)0, acc, false, false);
        }
        wait_async0();     // our async issues for the next buffer are done
        __syncthreads();   // everyone finished reading current buffer
    }
    return acc;
}

// forward step: x_hat = states_l @ E ; err = target - x_hat  (bf16 out)
__global__ __launch_bounds__(128)
void gemm_fwd(const __bf16* __restrict__ Abf, const __bf16* __restrict__ Ecm,
              const float* __restrict__ target, __bf16* __restrict__ errbf,
              int Kdim, int Nout) {
    extern __shared__ char smem[];
    const int lane = threadIdx.x & 31;
    const int wave = threadIdx.x >> 5;
    const int half = lane >> 4, r = lane & 15;
    const int mb = wave * 16;             // 4 waves cover batch rows 0..63
    const int n0 = blockIdx.x * 16;
    v8f acc = gemm_tile(Abf, Ecm, smem, Kdim, mb, n0, half, r);
    const int col = n0 + r;
#pragma unroll
    for (int v = 0; v < 8; ++v) {
        int row = mb + half * 8 + v;
        float e = target[row * Nout + col] - acc[v];
        errbf[row * Nout + col] = (__bf16)e;
    }
}

// update step: states_l += ALPHA * (err @ E^T) ; refresh bf16 shadow
__global__ __launch_bounds__(128)
void gemm_upd(const __bf16* __restrict__ Abf, const __bf16* __restrict__ Ecm,
              float* __restrict__ states, __bf16* __restrict__ sbf,
              int Kdim, int Nout) {
    extern __shared__ char smem[];
    const int lane = threadIdx.x & 31;
    const int wave = threadIdx.x >> 5;
    const int half = lane >> 4, r = lane & 15;
    const int mb = wave * 16;
    const int n0 = blockIdx.x * 16;
    v8f acc = gemm_tile(Abf, Ecm, smem, Kdim, mb, n0, half, r);
    const int col = n0 + r;
#pragma unroll
    for (int v = 0; v < 8; ++v) {
        int row = mb + half * 8 + v;
        float s = states[row * Nout + col] + ALPHA * acc[v];
        states[row * Nout + col] = s;
        sbf[row * Nout + col] = (__bf16)s;
    }
}

// ---------------------------------------------------------------------------
extern "C" void kernel_launch(void* const* d_in, const int* in_sizes, int n_in,
                              void* d_out, int out_size, void* d_ws, size_t ws_size,
                              hipStream_t stream) {
    const float* x  = (const float*)d_in[0];
    const float* w1 = (const float*)d_in[1];   // (256, 512, 8)
    const float* w2 = (const float*)d_in[2];   // (512, 256, 8)
    // n_iter (d_in[3]) is a fixed scalar (=20); loop count must be host-side.

    float* out0 = (float*)d_out;               // states[0]  (64, 2048)
    float* out1 = out0 + S0;                   // states[1]  (64, 4096)
    float* out2 = out1 + S1;                   // states[2]  (64, 2048)

    // workspace carve-out (all blocks are large powers of two -> aligned)
    char* ws = (char*)d_ws;
    __bf16* E1  = (__bf16*)ws; ws += (size_t)F0 * F1 * 2;    // fwd l=1 (2048 x 4096)
    __bf16* E1T = (__bf16*)ws; ws += (size_t)F1 * F0 * 2;    // upd l=1 (4096 x 2048)
    __bf16* E2  = (__bf16*)ws; ws += (size_t)F1 * F0 * 2;    // fwd l=2 (4096 x 2048)
    __bf16* E2T = (__bf16*)ws; ws += (size_t)F0 * F1 * 2;    // upd l=2 (2048 x 4096)
    __bf16* sb1 = (__bf16*)ws; ws += (size_t)BATCH * F1 * 2; // states1 bf16
    __bf16* sb2 = (__bf16*)ws; ws += (size_t)BATCH * F0 * 2; // states2 bf16
    __bf16* e1  = (__bf16*)ws; ws += (size_t)BATCH * F0 * 2; // err layer1 bf16
    __bf16* e2  = (__bf16*)ws; ws += (size_t)BATCH * F1 * 2; // err layer2 bf16

    init_states<<<S1 / 256, 256, 0, stream>>>(x, out0, out1, out2, sb1, sb2);

    const int EG = (F0 * F1) / 256;  // 32768 blocks; all four matrices same size
    // E1 [q=(m,k), m in D0][p=(n,a), n in D1] = s * w1[m][n][c]
    expand_weights<<<EG, 256, 0, stream>>>(w1, E1,  D1, D1 * KB, KB,      0);
    // E1T[q=(m',k), m' in D1][p=(n',a), n' in D0] = s*REV * w1[n'][m'][c]
    expand_weights<<<EG, 256, 0, stream>>>(w1, E1T, D0, KB,      D1 * KB, 1);
    // E2 [q in D1][p in D2] = s * w2[m][n][c]
    expand_weights<<<EG, 256, 0, stream>>>(w2, E2,  D0, D0 * KB, KB,      0);
    // E2T[q in D2][p in D1] = s*REV * w2[n'][m'][c]
    expand_weights<<<EG, 256, 0, stream>>>(w2, E2T, D1, KB,      D0 * KB, 1);

    for (int it = 0; it < N_ITER; ++it) {
        // layer 1: x_hat = s1 @ E1 ; err1 = states0 - x_hat
        gemm_fwd<<<F0 / 16, 128, LDSBYTES, stream>>>(sb1, E1, out0, e1, F1, F0);
        // states1 += alpha * err1 @ E1T
        gemm_upd<<<F1 / 16, 128, LDSBYTES, stream>>>(e1, E1T, out1, sb1, F0, F1);
        // layer 2: x_hat = s2 @ E2 ; err2 = states1 - x_hat
        gemm_fwd<<<F1 / 16, 128, LDSBYTES, stream>>>(sb2, E2, out1, e2, F0, F1);
        // states2 += alpha * err2 @ E2T
        gemm_upd<<<F0 / 16, 128, LDSBYTES, stream>>>(e2, E2T, out2, sb2, F1, F0);
    }
}
